// EquivWeightQueryModelOld_4114578670368
// MI455X (gfx1250) — matrix operand order repaired and checked by reading, not compiled, not run on hardware
//
#include <hip/hip_runtime.h>
#include <hip/hip_bf16.h>

// ---------------- model dims ----------------
#define NNODES 2048
#define NEDGE  32768
#define NQQ    32
#define DIMF   86
#define NPADF  96
#define SHD    16
#define NBR    10
#define DLIN   172
#define RMAXE  4.0f
#define FCUTF  4.0f

typedef __attribute__((ext_vector_type(16))) __bf16 v16bf;
typedef __attribute__((ext_vector_type(8)))  __bf16 v8bf;
typedef __attribute__((ext_vector_type(8)))  float  v8f;

__device__ inline __bf16 f2bf(float f) { return (__bf16)f; }  // hw cvt

__device__ inline void sh16(float x, float y, float z, float* o) {
  const float s3 = 1.7320508075688772f;
  const float c1 = 0.7905694150420949f;   // sqrt(5/8)
  const float c2 = 3.872983346207417f;    // sqrt(15)
  const float c3 = 0.6123724356957945f;   // sqrt(3/8)
  o[0] = 1.0f; o[1] = x; o[2] = y; o[3] = z;
  o[4] = s3*x*y; o[5] = s3*y*z; o[6] = 0.5f*(3.f*z*z-1.f); o[7] = s3*z*x;
  o[8] = 0.5f*s3*(x*x-y*y);
  o[9]  = c1*y*(3.f*x*x-y*y);
  o[10] = c2*x*y*z;
  o[11] = c3*y*(5.f*z*z-1.f);
  o[12] = 0.5f*z*(5.f*z*z-3.f);
  o[13] = c3*x*(5.f*z*z-1.f);
  o[14] = 0.5f*c2*z*(x*x-y*y);
  o[15] = c1*x*(x*x-3.f*y*y);
}

__device__ inline void rbf10(float r, float rmax, float* o) {
  float w = rmax / (NBR - 1);
  float t = r / rmax; t = t < 0.f ? 0.f : (t > 1.f ? 1.f : t);
  float env = 0.5f * (__cosf(3.14159265358979f * t) + 1.f) * (r < rmax ? 1.f : 0.f);
#pragma unroll
  for (int i = 0; i < NBR; ++i) {
    float d = (r - w * (float)i) / w;
    o[i] = __expf(-d * d) * env;
  }
}

__device__ inline int f2ord(float f) {       // order-preserving int encoding
  int i = __builtin_bit_cast(int, f);
  return (i >= 0) ? i : (i ^ 0x7fffffff);
}
__device__ inline float ord2f(int i) {
  int j = (i >= 0) ? i : (i ^ 0x7fffffff);
  return __builtin_bit_cast(float, j);
}

// ---- WMMA fragment loaders (bf16 A/B, contiguous k order shared by A and B) ----
// A 16x32: lane element i -> k = k0 + i + 16*lane_hi  (two 8-elem runs)
__device__ inline v16bf ld_afrag(const __bf16* ap, int k0, int lhi) {
  v8bf a0 = *(const v8bf*)(ap + k0 + (lhi << 4));
  v8bf a1 = *(const v8bf*)(ap + k0 + 8 + (lhi << 4));
  v16bf a;
#pragma unroll
  for (int i = 0; i < 8; ++i) { a[i] = a0[i]; a[i + 8] = a1[i]; }
  return a;
}
// B 32x16 from N-major weights: lane element i -> k = i (+16*lane_hi folded in bp)
__device__ inline v16bf ld_bfrag(const __bf16* bp) {
  v8bf b0 = *(const v8bf*)bp;
  v8bf b1 = *(const v8bf*)(bp + 8);
  v16bf b;
#pragma unroll
  for (int i = 0; i < 8; ++i) { b[i] = b0[i]; b[i + 8] = b1[i]; }
  return b;
}

// ---------------- fill kernels ----------------
__global__ void fillf_kernel(float* p, float v, int n) {
  int i = blockIdx.x * blockDim.x + threadIdx.x;
  if (i < n) p[i] = v;
}
__global__ void filli_kernel(int* p, int v, int n) {
  int i = blockIdx.x * blockDim.x + threadIdx.x;
  if (i < n) p[i] = v;
}

// ---------------- edge geometry: sh + rbf per edge ----------------
__global__ void edge_geom_kernel(const float* __restrict__ pos, const int* __restrict__ edge,
                                 float* __restrict__ shE, float* __restrict__ rbfE) {
  int e = blockIdx.x * blockDim.x + threadIdx.x;
  if (e >= NEDGE) return;
  int s = edge[e];
  int d = edge[NEDGE + e];
  float rx = pos[3*s]   - pos[3*d];
  float ry = pos[3*s+1] - pos[3*d+1];
  float rz = pos[3*s+2] - pos[3*d+2];
  float r = sqrtf(rx*rx + ry*ry + rz*rz + 1e-12f);
  float inv = 1.f / r;
  float o[16]; sh16(rx*inv, ry*inv, rz*inv, o);
#pragma unroll
  for (int i = 0; i < 16; ++i) shE[(size_t)e*16 + i] = o[i];
  float b[NBR]; rbf10(r, RMAXE, b);
#pragma unroll
  for (int i = 0; i < NBR; ++i) rbfE[(size_t)e*NBR + i] = b[i];
}

// ---------------- weight convert: f32 [Kin,Nin] -> bf16 [Npad][Kpad] (transposed+padded) ----
__global__ void convert_w_kernel(const float* __restrict__ W, __bf16* __restrict__ Wt,
                                 int Kin, int Nin, int Kpad, int Npad) {
  int i = blockIdx.x * blockDim.x + threadIdx.x;
  int tot = Kpad * Npad;
  if (i >= tot) return;
  int n = i / Kpad, k = i % Kpad;
  float v = (n < Nin && k < Kin) ? W[(size_t)k * Nin + n] : 0.f;
  Wt[(size_t)n * Kpad + k] = f2bf(v);
}

// ---------------- fused edge tensor-product GEMM (WMMA) ----------------
// out[e, o] = (sum_{f,s} feat[gidx[e], f] * shE[e, s] * W[f*16+s, o]) * (rbfE[e] @ R)[o]
__global__ __launch_bounds__(128)
void tp_gemm_kernel(const float* __restrict__ feat, int fdim,
                    const int* __restrict__ gidx,
                    const float* __restrict__ shE,
                    const float* __restrict__ rbfE,
                    const __bf16* __restrict__ Wt, int ktot,
                    const float* __restrict__ R,
                    float* __restrict__ out, int nout, int npad, int M) {
  __shared__ __bf16 s_A[32][1376];          // bf16 A panel (outer product), built once
  __shared__ float s_fs[32][DIMF + 2];
  __shared__ float s_sh[32][SHD];
  __shared__ float s_rb[32][NBR];
  __shared__ int   s_idx[32];
  int tid = threadIdx.x;
  int e0 = blockIdx.x * 32;

  if (tid < 32) {
    int e = e0 + tid;
    s_idx[tid] = (e < M) ? gidx[e] : 0;
  }
  __syncthreads();
  for (int i = tid; i < 32 * DIMF; i += 128) {
    int r = i / DIMF, c = i % DIMF;
    int e = e0 + r;
    s_fs[r][c] = (e < M && c < fdim) ? feat[(size_t)s_idx[r] * fdim + c] : 0.f;
  }
  for (int i = tid; i < 32 * SHD; i += 128) {
    int r = i / SHD, c = i % SHD;
    int e = e0 + r;
    s_sh[r][c] = (e < M) ? shE[(size_t)e * SHD + c] : 0.f;
  }
  for (int i = tid; i < 32 * NBR; i += 128) {
    int r = i / NBR, c = i % NBR;
    int e = e0 + r;
    s_rb[r][c] = (e < M) ? rbfE[(size_t)e * NBR + c] : 0.f;
  }
  __syncthreads();
  // build bf16 A panel: A[r][k] = fs[r][k>>4] * sh[r][k&15]  (each element converted once)
  for (int i = tid; i < 32 * ktot; i += 128) {
    int r = i / ktot, k = i - r * ktot;
    s_A[r][k] = f2bf(s_fs[r][k >> 4] * s_sh[r][k & 15]);
  }
  __syncthreads();

  int wave = tid >> 5, lane = tid & 31;
  int lhi = lane >> 4, lm = lane & 15;
  int mtile = wave & 1, nw = wave >> 1;
  int ntiles = npad >> 4;
  int mrow = mtile * 16 + lm;
  const __bf16* ap = &s_A[mrow][0];

  for (int nb = nw * 3; nb < ntiles; nb += 6) {
    const __bf16* bp[3];
#pragma unroll
    for (int j = 0; j < 3; ++j) {
      int nt = nb + j; if (nt > ntiles - 1) nt = ntiles - 1;   // clamp tail (no OOB)
      bp[j] = Wt + (size_t)(nt * 16 + lm) * ktot + (lhi << 4);
    }
    v8f acc[3] = {};
    for (int k0 = 0; k0 < ktot; k0 += 32) {
      __builtin_prefetch(bp[0] + k0 + 64, 0, 1);
      v16bf a = ld_afrag(ap, k0, lhi);
#pragma unroll
      for (int j = 0; j < 3; ++j) {
        v16bf b = ld_bfrag(bp[j] + k0);
        acc[j] = __builtin_amdgcn_wmma_f32_16x16x32_bf16(false, a, false, b,
                                                         (short)0, acc[j], false, false);
      }
    }
#pragma unroll
    for (int j = 0; j < 3; ++j) {
      int nt = nb + j;
      if (nt < ntiles) {
#pragma unroll
        for (int r = 0; r < 8; ++r) {
          int m = mtile * 16 + r + (lhi << 3);
          int n = nt * 16 + lm;
          int e = e0 + m;
          if (e < M && n < nout) {
            float rad = 1.f;
            if (R) {
              rad = 0.f;
#pragma unroll
              for (int jj = 0; jj < NBR; ++jj) rad += s_rb[m][jj] * R[(size_t)jj * nout + n];
            }
            out[(size_t)e * nout + n] = acc[j][r] * rad;
          }
        }
      }
    }
  }
}

// ---------------- dense GEMM (WMMA): C[m,n] = A[m,:kin] * W + addsrc ----------------
__global__ __launch_bounds__(128)
void dense_gemm_kernel(const float* __restrict__ A, int kin, int kpad,
                       const __bf16* __restrict__ Wt,   // [npad][kpad] bf16 N-major
                       const float* __restrict__ addsrc,
                       float* __restrict__ out, int nout, int npad, int M) {
  __shared__ __bf16 s_a[32][200];   // kpad <= 192, row stride 400B (16B-aligned)
  int tid = threadIdx.x;
  int m0 = blockIdx.x * 32;
  for (int i = tid; i < 32 * kpad; i += 128) {
    int r = i / kpad, c = i - r * kpad;
    int m = m0 + r;
    s_a[r][c] = f2bf((m < M && c < kin) ? A[(size_t)m * kin + c] : 0.f);
  }
  __syncthreads();
  int wave = tid >> 5, lane = tid & 31;
  int lhi = lane >> 4, lm = lane & 15;
  int mtile = wave & 1, nw = wave >> 1;
  int ntiles = npad >> 4;
  int mrow = mtile * 16 + lm;
  const __bf16* ap = &s_a[mrow][0];

  for (int nb = nw * 3; nb < ntiles; nb += 6) {
    const __bf16* bp[3];
#pragma unroll
    for (int j = 0; j < 3; ++j) {
      int nt = nb + j; if (nt > ntiles - 1) nt = ntiles - 1;
      bp[j] = Wt + (size_t)(nt * 16 + lm) * kpad + (lhi << 4);
    }
    v8f acc[3] = {};
    for (int k0 = 0; k0 < kpad; k0 += 32) {
      __builtin_prefetch(bp[0] + k0 + 64, 0, 1);
      v16bf a = ld_afrag(ap, k0, lhi);
#pragma unroll
      for (int j = 0; j < 3; ++j) {
        v16bf b = ld_bfrag(bp[j] + k0);
        acc[j] = __builtin_amdgcn_wmma_f32_16x16x32_bf16(false, a, false, b,
                                                         (short)0, acc[j], false, false);
      }
    }
#pragma unroll
    for (int j = 0; j < 3; ++j) {
      int nt = nb + j;
      if (nt < ntiles) {
#pragma unroll
        for (int r = 0; r < 8; ++r) {
          int m = m0 + mtile * 16 + r + (lhi << 3);
          int n = nt * 16 + lm;
          if (m < M && n < nout) {
            float v = acc[j][r];
            if (addsrc) v += addsrc[(size_t)m * nout + n];
            out[(size_t)m * nout + n] = v;
          }
        }
      }
    }
  }
}

// ---------------- attention: logit + segment max ----------------
__global__ void attn_logit_kernel(const float* __restrict__ q, const float* __restrict__ k,
                                  const int* __restrict__ edge,
                                  float* __restrict__ logit, int* __restrict__ segmax) {
  int e = blockIdx.x * blockDim.x + threadIdx.x;
  if (e >= NEDGE) return;
  int d = edge[NEDGE + e];
  float s = 0.f;
#pragma unroll 2
  for (int i = 0; i < DIMF; ++i) s += q[(size_t)d * DIMF + i] * k[(size_t)e * DIMF + i];
  s *= 0.10783277320343841f;   // 1/sqrt(86)
  logit[e] = s;
  atomicMax(&segmax[d], f2ord(s));
}

__global__ void attn_exp_kernel(const float* __restrict__ logit, const int* __restrict__ edge,
                                const int* __restrict__ segmax,
                                float* __restrict__ ex, float* __restrict__ denom) {
  int e = blockIdx.x * blockDim.x + threadIdx.x;
  if (e >= NEDGE) return;
  int d = edge[NEDGE + e];
  float x = __expf(logit[e] - ord2f(segmax[d]));
  ex[e] = x;
  atomicAdd(&denom[d], x);
}

__global__ void attn_scatter_kernel(const float* __restrict__ ex, const float* __restrict__ denom,
                                    const float* __restrict__ v, const int* __restrict__ edge,
                                    float* __restrict__ agg) {
  int i = blockIdx.x * blockDim.x + threadIdx.x;
  int tot = NEDGE * DIMF;
  if (i >= tot) return;
  int e = i / DIMF, o = i - e * DIMF;
  int d = edge[NEDGE + e];
  float a = ex[e] / (denom[d] + 1e-9f);
  atomicAdd(&agg[(size_t)d * DIMF + o], a * v[(size_t)e * DIMF + o]);
}

// ---------------- equivariant layernorm ----------------
__global__ void eln_kernel(const float* __restrict__ x, const float* __restrict__ g,
                           float* __restrict__ out, int M) {
  int m = blockIdx.x * blockDim.x + threadIdx.x;
  if (m >= M) return;
  const int mult[4] = {10, 10, 5, 3};
  int off = 0, go = 0;
  for (int b = 0; b < 4; ++b) {
    int l = b, d = 2 * l + 1, mm = mult[b];
    float mean = 0.f;
    if (l == 0) {
      for (int i = 0; i < mm; ++i) mean += x[(size_t)m * DIMF + off + i];
      mean /= (float)mm;
    }
    float ss = 0.f;
    for (int i = 0; i < mm * d; ++i) {
      float v = x[(size_t)m * DIMF + off + i] - (l == 0 ? mean : 0.f);
      ss += v * v;
    }
    float rms = sqrtf(ss / (float)(mm * d) + 1e-6f);
    for (int i = 0; i < mm * d; ++i) {
      float v = x[(size_t)m * DIMF + off + i] - (l == 0 ? mean : 0.f);
      out[(size_t)m * DIMF + off + i] = v / rms * g[go + i / d];
    }
    off += mm * d; go += mm;
  }
}

// ---------------- tensor-field reduction (factorized): T[n, s*nout+o] ----------------
__global__ __launch_bounds__(128)
void field_reduce_kernel(const float* __restrict__ T, const float* __restrict__ R,
                         const float* __restrict__ pos, const float* __restrict__ qpos,
                         float* __restrict__ out, int nout, int Q) {
  __shared__ float s_acc[96];
  __shared__ int s_cnt;
  int q = blockIdx.x;
  if (q >= Q) return;
  int tid = threadIdx.x;
  if (tid < 96) s_acc[tid] = 0.f;
  if (tid == 0) s_cnt = 0;
  __syncthreads();
  float qx = qpos[3*q], qy = qpos[3*q+1], qz = qpos[3*q+2];
  int lc = 0;
  for (int n = tid; n < NNODES; n += 128) {
    float rx = pos[3*n] - qx, ry = pos[3*n+1] - qy, rz = pos[3*n+2] - qz;
    float r = sqrtf(rx*rx + ry*ry + rz*rz + 1e-12f);
    float inv = 1.f / r;
    float sh[16]; sh16(rx*inv, ry*inv, rz*inv, sh);
    float rb[NBR]; rbf10(r, FCUTF, rb);
    if (r < FCUTF) lc++;
    const float* Tn = T + (size_t)n * 16 * nout;
    for (int o = 0; o < nout; ++o) {
      float mv = 0.f;
#pragma unroll
      for (int s = 0; s < 16; ++s) mv += sh[s] * Tn[(size_t)s * nout + o];
      float rad = 0.f;
#pragma unroll
      for (int j = 0; j < NBR; ++j) rad += rb[j] * R[(size_t)j * nout + o];
      atomicAdd(&s_acc[o], mv * rad);
    }
  }
  atomicAdd(&s_cnt, lc);
  __syncthreads();
  if (tid < nout) {
    float c = (float)(s_cnt < 1 ? 1 : s_cnt);
    out[(size_t)q * nout + tid] = s_acc[tid] / c;
  }
}

// ---------------- weight-field finalize: LN(10) + linear -> scalar ----------------
__global__ void wf_final_kernel(const float* __restrict__ f, const float* __restrict__ g,
                                const float* __restrict__ b, const float* __restrict__ lw,
                                const float* __restrict__ lb, float* __restrict__ logit, int Q) {
  int q = blockIdx.x * blockDim.x + threadIdx.x;
  if (q >= Q) return;
  float mean = 0.f;
  for (int i = 0; i < 10; ++i) mean += f[(size_t)q * 10 + i];
  mean *= 0.1f;
  float var = 0.f;
  for (int i = 0; i < 10; ++i) { float d = f[(size_t)q * 10 + i] - mean; var += d * d; }
  var *= 0.1f;
  float inv = rsqrtf(var + 1e-6f);
  float s = 0.f;
  for (int i = 0; i < 10; ++i)
    s += ((f[(size_t)q * 10 + i] - mean) * inv * g[i] + b[i]) * lw[i];
  logit[q] = s + lb[0];
}

// ---------------- top-32 selection (single block) ----------------
__global__ void topk_kernel(const float* __restrict__ logit, int* __restrict__ idx_out,
                            float* __restrict__ qp_out, const float* __restrict__ pos) {
  __shared__ float s_v[256];
  __shared__ int s_i[256];
  __shared__ unsigned char taken[NNODES];
  int tid = threadIdx.x;
  for (int i = tid; i < NNODES; i += 256) taken[i] = 0;
  __syncthreads();
  for (int it = 0; it < NQQ; ++it) {
    float bv = -3.4e38f; int bi = 0;
    for (int n = tid; n < NNODES; n += 256) {
      if (!taken[n] && logit[n] > bv) { bv = logit[n]; bi = n; }
    }
    s_v[tid] = bv; s_i[tid] = bi;
    __syncthreads();
    for (int st = 128; st > 0; st >>= 1) {
      if (tid < st) {
        if (s_v[tid + st] > s_v[tid] ||
            (s_v[tid + st] == s_v[tid] && s_i[tid + st] < s_i[tid])) {
          s_v[tid] = s_v[tid + st]; s_i[tid] = s_i[tid + st];
        }
      }
      __syncthreads();
    }
    if (tid == 0) {
      int w = s_i[0];
      idx_out[it] = w; taken[w] = 1;
      qp_out[3*it]   = pos[3*w];
      qp_out[3*it+1] = pos[3*w+1];
      qp_out[3*it+2] = pos[3*w+2];
    }
    __syncthreads();
  }
}

__global__ void softmax32_kernel(const float* __restrict__ logit, float* __restrict__ qa) {
  __shared__ float s[32];
  int t = threadIdx.x;
  s[t] = logit[t];
  __syncthreads();
  float m = s[0];
  for (int i = 1; i < 32; ++i) m = fmaxf(m, s[i]);
  float e = __expf(s[t] - m);
  __shared__ float se[32];
  se[t] = e;
  __syncthreads();
  float d = 0.f;
  for (int i = 0; i < 32; ++i) d += se[i];
  qa[t] = e / d;
}

// ================= host orchestration =================
extern "C" void kernel_launch(void* const* d_in, const int* in_sizes, int n_in,
                              void* d_out, int out_size, void* d_ws, size_t ws_size,
                              hipStream_t stream) {
  (void)in_sizes; (void)n_in; (void)out_size; (void)ws_size;
  const float* feature = (const float*)d_in[0];
  const float* pos     = (const float*)d_in[1];
  const int*   edge    = (const int*)d_in[2];
  auto P = [&](int i) { return (const float*)d_in[i]; };

  // ---- workspace carve ----
  char* w = (char*)d_ws;
  auto alloc = [&](size_t bytes) {
    void* p = (void*)w;
    w += (bytes + 255) & ~((size_t)255);
    return p;
  };
  float*  shE    = (float*)alloc((size_t)NEDGE * 16 * 4);
  float*  rbfE   = (float*)alloc((size_t)NEDGE * NBR * 4);
  float*  kvec   = (float*)alloc((size_t)NEDGE * DIMF * 4);
  float*  vvec   = (float*)alloc((size_t)NEDGE * DIMF * 4);
  float*  logitE = (float*)alloc((size_t)NEDGE * 4);
  float*  exE    = (float*)alloc((size_t)NEDGE * 4);
  float*  denom  = (float*)alloc((size_t)NNODES * 4);
  int*    segmax = (int*)alloc((size_t)NNODES * 4);
  float*  agg    = (float*)alloc((size_t)NNODES * DIMF * 4);
  float*  qbuf   = (float*)alloc((size_t)NNODES * DIMF * 4);
  float*  tmp1   = (float*)alloc((size_t)NNODES * DIMF * 4);
  float*  tmp2   = (float*)alloc((size_t)NNODES * DLIN * 4);
  float*  tmp3   = (float*)alloc((size_t)NNODES * DIMF * 4);
  float*  featA  = (float*)alloc((size_t)NNODES * DIMF * 4);
  float*  featB  = (float*)alloc((size_t)NNODES * DIMF * 4);
  float*  Twf    = (float*)alloc((size_t)NNODES * 160 * 4);
  float*  T2     = (float*)alloc((size_t)NNODES * 1376 * 4);
  float*  fqN    = (float*)alloc((size_t)NNODES * 10 * 4);
  float*  nlogit = (float*)alloc((size_t)NNODES * 4);
  int*    idx32  = (int*)alloc(32 * 4);
  float*  fq32   = (float*)alloc(32 * 10 * 4);
  float*  qlog32 = (float*)alloc(32 * 4);
  float*  f32buf = (float*)alloc(32 * DIMF * 4);
  float*  ones28 = (float*)alloc(32 * 4);
  __bf16* Wt     = (__bf16*)alloc((size_t)132096 * 2 + 1024);

  float* out_qf = (float*)d_out;             // [32*86]
  float* out_qp = out_qf + 32 * DIMF;        // [32*3]
  float* out_qa = out_qp + 32 * 3;           // [32]

  auto conv = [&](const float* Wp, int Kin, int Nin, int Kpad, int Npad) {
    int tot = Kpad * Npad;
    convert_w_kernel<<<(tot + 255) / 256, 256, 0, stream>>>(Wp, Wt, Kin, Nin, Kpad, Npad);
  };

  // ---- edge geometry (sh + rbf) ----
  edge_geom_kernel<<<NEDGE / 256, 256, 0, stream>>>(pos, edge, shE, rbfE);
  fillf_kernel<<<1, 32, 0, stream>>>(ones28, 1.0f, 28);

  const int* srcIdx = edge;   // edge[0] row = src indices

  // ---- 7 SE(3)-transformer layers ----
  const float* feat = feature;
  for (int i = 0; i < 7; ++i) {
    int base = 3 + 9 * i;
    const float* Wq  = P(base + 0);
    const float* Wk  = P(base + 1);
    const float* Rk  = P(base + 2);
    const float* Wv  = P(base + 3);
    const float* Rv  = P(base + 4);
    const float* Wsi = P(base + 5);
    const float* Wla = P(base + 6);
    const float* Wlb = P(base + 7);
    const float* gg  = P(base + 8);

    int fdim  = (i == 0) ? 3 : DIMF;
    int ktot  = (i == 0) ? 64 : (DIMF * 16);     // padded K for TP GEMM
    int kinTP = fdim * 16;
    int kpadq = (i == 0) ? 32 : NPADF;

    // q = feat @ Wq
    conv(Wq, fdim, DIMF, kpadq, NPADF);
    dense_gemm_kernel<<<NNODES / 32, 128, 0, stream>>>(feat, fdim, kpadq, Wt, nullptr,
                                                       qbuf, DIMF, NPADF, NNODES);
    // k = TP(feat[src], sh) * (rbf@Rk)
    conv(Wk, kinTP, DIMF, ktot, NPADF);
    tp_gemm_kernel<<<NEDGE / 32, 128, 0, stream>>>(feat, fdim, srcIdx, shE, rbfE,
                                                   Wt, ktot, Rk, kvec, DIMF, NPADF, NEDGE);
    // v = TP(feat[src], sh) * (rbf@Rv)
    conv(Wv, kinTP, DIMF, ktot, NPADF);
    tp_gemm_kernel<<<NEDGE / 32, 128, 0, stream>>>(feat, fdim, srcIdx, shE, rbfE,
                                                   Wt, ktot, Rv, vvec, DIMF, NPADF, NEDGE);
    // softmax-scatter attention
    filli_kernel<<<NNODES / 256, 256, 0, stream>>>(segmax, (int)0x80000000, NNODES);
    fillf_kernel<<<NNODES / 256, 256, 0, stream>>>(denom, 0.f, NNODES);
    fillf_kernel<<<(NNODES * DIMF + 255) / 256, 256, 0, stream>>>(agg, 0.f, NNODES * DIMF);
    attn_logit_kernel<<<NEDGE / 256, 256, 0, stream>>>(qbuf, kvec, edge, logitE, segmax);
    attn_exp_kernel<<<NEDGE / 256, 256, 0, stream>>>(logitE, edge, segmax, exE, denom);
    attn_scatter_kernel<<<(NEDGE * DIMF + 255) / 256, 256, 0, stream>>>(exE, denom, vvec,
                                                                        edge, agg);
    // out = agg + feat@Wsi ; out = (out@Wla)@Wlb (+ feat skip) ; eln
    conv(Wsi, fdim, DIMF, kpadq, NPADF);
    dense_gemm_kernel<<<NNODES / 32, 128, 0, stream>>>(feat, fdim, kpadq, Wt, agg,
                                                       tmp1, DIMF, NPADF, NNODES);
    conv(Wla, DIMF, DLIN, NPADF, 176);
    dense_gemm_kernel<<<NNODES / 32, 128, 0, stream>>>(tmp1, DIMF, NPADF, Wt, nullptr,
                                                       tmp2, DLIN, 176, NNODES);
    conv(Wlb, DLIN, DIMF, 192, NPADF);
    dense_gemm_kernel<<<NNODES / 32, 128, 0, stream>>>(tmp2, DLIN, 192, Wt,
                                                       (i > 0) ? feat : nullptr,
                                                       tmp3, DIMF, NPADF, NNODES);
    float* nf = (i & 1) ? featB : featA;
    eln_kernel<<<NNODES / 128, 128, 0, stream>>>(tmp3, gg, nf, NNODES);
    feat = nf;
  }

  const float* wfW = P(66), * wfR = P(67), * wfg = P(68), * wfb = P(69);
  const float* wflw = P(70), * wflb = P(71);
  const float* ffW = P(72), * ffR = P(73);

  // ---- weight field over all nodes: T = feat @ wfW_flat [86 -> 160] ----
  conv(wfW, DIMF, 160, NPADF, 160);
  dense_gemm_kernel<<<NNODES / 32, 128, 0, stream>>>(feat, DIMF, NPADF, Wt, nullptr,
                                                     Twf, 160, 160, NNODES);
  field_reduce_kernel<<<NNODES, 128, 0, stream>>>(Twf, wfR, pos, pos, fqN, 10, NNODES);
  wf_final_kernel<<<NNODES / 256, 256, 0, stream>>>(fqN, wfg, wfb, wflw, wflb, nlogit, NNODES);
  topk_kernel<<<1, 256, 0, stream>>>(nlogit, idx32, out_qp, pos);

  // ---- ff field at 32 queries: T2 = feat @ ffW_flat [86 -> 1376] ----
  conv(ffW, DIMF, 1376, NPADF, 1376);
  dense_gemm_kernel<<<NNODES / 32, 128, 0, stream>>>(feat, DIMF, NPADF, Wt, nullptr,
                                                     T2, 1376, 1376, NNODES);
  field_reduce_kernel<<<32, 128, 0, stream>>>(T2, ffR, pos, out_qp, f32buf, DIMF, 32);
  eln_kernel<<<1, 32, 0, stream>>>(f32buf, ones28, out_qf, 32);

  // ---- qa = softmax(weight_field(qp)) ----
  field_reduce_kernel<<<32, 128, 0, stream>>>(Twf, wfR, pos, out_qp, fq32, 10, 32);
  wf_final_kernel<<<1, 32, 0, stream>>>(fq32, wfg, wfb, wflw, wflb, qlog32, 32);
  softmax32_kernel<<<1, 32, 0, stream>>>(qlog32, out_qa);
}
